// SparseVoxelEncoder4x_70300024701153
// MI455X (gfx1250) — matrix operand order repaired and checked by reading, not compiled, not run on hardware
//
#include <hip/hip_runtime.h>

typedef unsigned short u16;
typedef __attribute__((ext_vector_type(16))) __bf16 v16bf;
typedef __attribute__((ext_vector_type(8)))  float  v8f;

// ---------- bf16 helpers (RNE) ----------
__device__ __forceinline__ u16 f2bf(float f) {
  unsigned u = __float_as_uint(f);
  u += 0x7FFFu + ((u >> 16) & 1u);
  return (u16)(u >> 16);
}
__device__ __forceinline__ float bf2f(u16 h) {
  return __uint_as_float(((unsigned)h) << 16);
}

// ---------- zero fill ----------
__global__ void k_zero(uint4* p, long n) {
  long i = (long)blockIdx.x * blockDim.x + threadIdx.x;
  long stride = (long)gridDim.x * blockDim.x;
  uint4 z = make_uint4(0u, 0u, 0u, 0u);
  for (; i < n; i += stride) p[i] = z;
}

// ---------- scatter voxels -> padded dense bf16 [z+1][y+1][x+1][16] + unpadded mask ----------
__global__ void k_scatter(const float* __restrict__ vf, const int* __restrict__ coors,
                          int n, u16* __restrict__ x16, float* __restrict__ mask,
                          int H, int W, int Hp, int Wp) {
  int i = blockIdx.x * blockDim.x + threadIdx.x;
  if (i >= n) return;
  int z = coors[i * 4 + 1];
  int y = coors[i * 4 + 2];
  int x = coors[i * 4 + 3];
  long pp = ((long)(z + 1) * Hp + (y + 1)) * Wp + (x + 1);
  u16* o = x16 + pp * 16;
#pragma unroll
  for (int c = 0; c < 16; ++c) o[c] = f2bf(vf[(long)i * 16 + c]);
  mask[((long)z * H + y) * W + x] = 1.0f;
}

// ---------- pack weights [Cout][Cin][27] fp32 -> WMMA A-layout bf16 (zero-padded K) ----------
__global__ void k_pack_w(const float* __restrict__ w, u16* __restrict__ out,
                         int cout, int cin, int kc) {
  long total = (long)(cout / 16) * kc * 512;
  long i = (long)blockIdx.x * blockDim.x + threadIdx.x;
  if (i >= total) return;
  int e = (int)(i & 15);
  int lane = (int)((i >> 4) & 31);
  long t = i >> 9;             // mt*kc + kk
  int kk = (int)(t % kc);
  int mt = (int)(t / kc);
  int h = lane >> 4, l16 = lane & 15;
  int koff = e + 16 * h;   // contiguous k per lane half (matches B loads)
  int K = kk * 32 + koff;
  u16 v = 0;
  if (K < 27 * cin) {
    int tap = K / cin;
    int ci = K - tap * cin;
    int co = mt * 16 + l16;
    v = f2bf(w[((long)co * cin + ci) * 27 + tap]);
  }
  out[i] = v;
}

// ---------- implicit-GEMM conv3d 3x3x3 pad1 over halo-padded bf16 input ----------
// x: bf16 [Dp][Hp][Wp][cin] channels-last, halo zeroed; y: f32 [P][cout] dense
__global__ __launch_bounds__(128)
void k_conv3d_wmma(const u16* __restrict__ x, const u16* __restrict__ wp,
                   float* __restrict__ y,
                   int cin, int cinShift, int cout, int kc, int stride,
                   int Hp, int Wp, int Hout, int Wout, int P) {
  const int lane = threadIdx.x & 31;
  const int wave = threadIdx.x >> 5;
  const int half = lane >> 4;
  const int l16 = lane & 15;
  const int mt = blockIdx.y * 2;          // two 16-row co tiles
  const int HW = Hout * Wout;

  int p0 = blockIdx.x * 128 + wave * 32 + l16;
  int p1 = p0 + 16;
  bool ok0 = p0 < P, ok1 = p1 < P;
  int q0 = ok0 ? p0 : P - 1, q1 = ok1 ? p1 : P - 1;

  int oz0 = q0 / HW, r0 = q0 - oz0 * HW, oy0 = r0 / Wout, ox0 = r0 - oy0 * Wout;
  int oz1 = q1 / HW, r1 = q1 - oz1 * HW, oy1 = r1 / Wout, ox1 = r1 - oy1 * Wout;
  // tap(0,0,0) reads padded coord (oz*s, oy*s, ox*s) -- always in-bounds
  int pb0 = (((oz0 * stride) * Hp + oy0 * stride) * Wp + ox0 * stride) * cin;
  int pb1 = (((oz1 * stride) * Hp + oy1 * stride) * Wp + ox1 * stride) * cin;

  v8f acc00 = {}, acc01 = {}, acc10 = {}, acc11 = {};

  const u16* wpa = wp + ((long)mt * kc * 32 + lane) * 16;
  const u16* wpb = wpa + (long)kc * 512;
  int koff = half << 4;
  for (int kk = 0; kk < kc; ++kk, koff += 32) {
    int tap = koff >> cinShift;
    int ci0 = koff & (cin - 1);
    tap = tap < 27 ? tap : 26;  // K-padding chunks: packed weights are zero there
    int kd = tap / 9, rr = tap - kd * 9, kh = rr / 3, kw = rr - kh * 3;
    int toff = ((kd * Hp + kh) * Wp + kw) * cin + ci0;
    v16bf b0 = *(const v16bf*)(x + pb0 + toff);
    v16bf b1 = *(const v16bf*)(x + pb1 + toff);
    v16bf a0 = *(const v16bf*)(wpa + (long)kk * 512);
    v16bf a1 = *(const v16bf*)(wpb + (long)kk * 512);
    acc00 = __builtin_amdgcn_wmma_f32_16x16x32_bf16(false, a0, false, b0, (short)0, acc00, false, false);
    acc01 = __builtin_amdgcn_wmma_f32_16x16x32_bf16(false, a0, false, b1, (short)0, acc01, false, false);
    acc10 = __builtin_amdgcn_wmma_f32_16x16x32_bf16(false, a1, false, b0, (short)0, acc10, false, false);
    acc11 = __builtin_amdgcn_wmma_f32_16x16x32_bf16(false, a1, false, b1, (short)0, acc11, false, false);
  }

  // C/D layout: VGPR r -> (co = tile + r + 8*half, pos = p_group + l16)
  int coOff = mt * 16 + 8 * half;
  if (ok0) {
    *(v8f*)(y + (long)p0 * cout + coOff) = acc00;
    *(v8f*)(y + (long)p0 * cout + coOff + 16) = acc10;
  }
  if (ok1) {
    *(v8f*)(y + (long)p1 * cout + coOff) = acc01;
    *(v8f*)(y + (long)p1 * cout + coOff + 16) = acc11;
  }
}

// ---------- masked-BN stats ----------
__global__ __launch_bounds__(256)
void k_bn_stats(const float* __restrict__ y, const float* __restrict__ mask,
                float* __restrict__ sums, int C, int cShift, long P) {
  __shared__ float s[257];
  int t = threadIdx.x;
  int twoC = 2 * C;
  for (int i = t; i < twoC + 1; i += 256) s[i] = 0.f;
  __syncthreads();
  int c = t & (C - 1);               // C | 256, stride multiple of 256
  long total = P << cShift;
  long stride = (long)gridDim.x * 256;
  float s1 = 0.f, s2 = 0.f, cm = 0.f;
  for (long i = (long)blockIdx.x * 256 + t; i < total; i += stride) {
    float m = mask[i >> cShift];
    float v = y[i] * m;
    s1 += v;
    s2 += v * v;
    if (c == 0) cm += m;
  }
  atomicAdd(&s[c], s1);
  atomicAdd(&s[C + c], s2);
  if (c == 0) atomicAdd(&s[twoC], cm);
  __syncthreads();
  for (int i = t; i < twoC + 1; i += 256) unsafeAtomicAdd(&sums[i], s[i]);
}

// ---------- masked-BN apply (+optional residual) + relu -> padded bf16 ----------
__global__ void k_bn_apply(const float* __restrict__ y, const float* __restrict__ sums,
                           const float* __restrict__ gamma, const float* __restrict__ beta,
                           const float* __restrict__ mask, const u16* __restrict__ res,
                           u16* __restrict__ out, int C, int cShift, long P,
                           int H, int W, int Hp, int Wp) {
  long total = P << cShift;
  long i = (long)blockIdx.x * blockDim.x + threadIdx.x;
  if (i >= total) return;
  int c = (int)(i & (C - 1));
  long p = i >> cShift;
  int z = (int)(p / ((long)H * W));
  int rem = (int)(p - (long)z * H * W);
  int yy = rem / W, xx = rem - yy * W;
  long pp = ((long)(z + 1) * Hp + (yy + 1)) * Wp + (xx + 1);
  float cnt = sums[2 * C];
  float mean = sums[c] / cnt;
  float var = sums[C + c] / cnt - mean * mean;
  float inv = rsqrtf(var + 1e-5f);
  float xn = (y[i] - mean) * inv * gamma[c] + beta[c];
  float v = xn * mask[p];
  if (res) v += bf2f(res[pp * C + c]);
  out[pp * C + c] = f2bf(fmaxf(v, 0.f));
}

// ---------- per-voxel GN (stats over cg channels) + mask + relu -> padded bf16 ----------
__global__ void k_gn_apply_bf16(const float* __restrict__ y, const float* __restrict__ gamma,
                                const float* __restrict__ beta, const float* __restrict__ mask,
                                u16* __restrict__ out, int C, int cg, long P,
                                int H, int W, int Hp, int Wp) {
  int G = C / cg;
  long total = P * G;
  long i = (long)blockIdx.x * blockDim.x + threadIdx.x;
  if (i >= total) return;
  long p = i / G;
  int g = (int)(i - p * G);
  int z = (int)(p / ((long)H * W));
  int rem = (int)(p - (long)z * H * W);
  int yy = rem / W, xx = rem - yy * W;
  long pp = ((long)(z + 1) * Hp + (yy + 1)) * Wp + (xx + 1);
  const float* yp = y + p * C + (long)g * cg;
  float mean = 0.f;
  for (int j = 0; j < cg; ++j) mean += yp[j];
  mean /= (float)cg;
  float var = 0.f;
  for (int j = 0; j < cg; ++j) { float d = yp[j] - mean; var += d * d; }
  var /= (float)cg;
  float inv = rsqrtf(var + 1e-5f);
  float m = mask[p];
  for (int j = 0; j < cg; ++j) {
    int c = g * cg + j;
    float v = ((yp[j] - mean) * inv * gamma[c] + beta[c]) * m;
    out[pp * C + c] = f2bf(fmaxf(v, 0.f));
  }
}

// ---------- final GN (C=128, cg=8) + mask + relu, write transposed (C,W,H,D) fp32 ----------
__global__ void k_gn_out(const float* __restrict__ y, const float* __restrict__ gamma,
                         const float* __restrict__ beta, const float* __restrict__ mask,
                         float* __restrict__ out, int Dd, int Hh, int Ww) {
  long P = (long)Dd * Hh * Ww;
  long total = P * 16;
  long i = (long)blockIdx.x * blockDim.x + threadIdx.x;
  if (i >= total) return;
  long p = i >> 4;
  int g = (int)(i & 15);
  int z = (int)(p / ((long)Hh * Ww));
  int rem = (int)(p - (long)z * Hh * Ww);
  int yy = rem / Ww, xx = rem - yy * Ww;
  const float* yp = y + p * 128 + g * 8;
  float mean = 0.f;
  for (int j = 0; j < 8; ++j) mean += yp[j];
  mean *= 0.125f;
  float var = 0.f;
  for (int j = 0; j < 8; ++j) { float d = yp[j] - mean; var += d * d; }
  var *= 0.125f;
  float inv = rsqrtf(var + 1e-5f);
  float m = mask[p];
  for (int j = 0; j < 8; ++j) {
    int c = g * 8 + j;
    float v = ((yp[j] - mean) * inv * gamma[c] + beta[c]) * m;
    out[(((long)c * Ww + xx) * Hh + yy) * Dd + z] = fmaxf(v, 0.f);
  }
}

// ---------- mask downsample: ones 3x3x3 stride2 pad1, >0 (unpadded masks) ----------
__global__ void k_down_mask(const float* __restrict__ mi, float* __restrict__ mo,
                            int Di, int Hi, int Wi, int Do, int Ho, int Wo) {
  long P = (long)Do * Ho * Wo;
  long i = (long)blockIdx.x * blockDim.x + threadIdx.x;
  if (i >= P) return;
  int z = (int)(i / ((long)Ho * Wo));
  int rem = (int)(i - (long)z * Ho * Wo);
  int y = rem / Wo, x = rem - y * Wo;
  float s = 0.f;
  for (int kd = 0; kd < 3; ++kd)
    for (int kh = 0; kh < 3; ++kh)
      for (int kw = 0; kw < 3; ++kw) {
        int iz = z * 2 + kd - 1, iy = y * 2 + kh - 1, ix = x * 2 + kw - 1;
        if ((unsigned)iz < (unsigned)Di && (unsigned)iy < (unsigned)Hi &&
            (unsigned)ix < (unsigned)Wi)
          s += mi[((long)iz * Hi + iy) * Wi + ix];
      }
  mo[i] = s > 0.f ? 1.f : 0.f;
}

// =======================================================================
static inline size_t align256(size_t x) { return (x + 255) & ~(size_t)255; }

static void run_conv(hipStream_t s, const u16* act, const u16* wp, float* y,
                     int cin, int cout, int stride, int Hp, int Wp,
                     int Do, int Ho, int Wo) {
  int P = Do * Ho * Wo;
  int kc = (27 * cin + 31) / 32;
  int cinShift = __builtin_ctz(cin);
  dim3 grid((P + 127) / 128, cout / 32);
  k_conv3d_wmma<<<grid, 128, 0, s>>>(act, wp, y, cin, cinShift, cout, kc, stride,
                                     Hp, Wp, Ho, Wo, P);
}

static void run_bn(hipStream_t s, const float* y, float* sums, const float* g,
                   const float* b, const float* mask, const u16* res, u16* out,
                   int C, int Dd, int Hh, int Ww, int Hp, int Wp) {
  long P = (long)Dd * Hh * Ww;
  int cShift = __builtin_ctz(C);
  k_zero<<<1, 256, 0, s>>>((uint4*)sums, 256);  // 4KB
  long total = P << cShift;
  int sblocks = (int)((total + 255) / 256);
  if (sblocks > 960) sblocks = 960;
  k_bn_stats<<<sblocks, 256, 0, s>>>(y, mask, sums, C, cShift, P);
  k_bn_apply<<<(int)((total + 255) / 256), 256, 0, s>>>(y, sums, g, b, mask, res, out,
                                                        C, cShift, P, Hh, Ww, Hp, Wp);
}

static void run_pack(hipStream_t s, const float* w, u16* out, int cout, int cin) {
  int kc = (27 * cin + 31) / 32;
  long total = (long)(cout / 16) * kc * 512;
  k_pack_w<<<(int)((total + 255) / 256), 256, 0, s>>>(w, out, cout, cin, kc);
}

extern "C" void kernel_launch(void* const* d_in, const int* in_sizes, int n_in,
                              void* d_out, int out_size, void* d_ws, size_t ws_size,
                              hipStream_t stream) {
  const float* vf = (const float*)d_in[0];
  const int* coors = (const int*)d_in[1];
  const int N = in_sizes[0] / 16;
#define FP(i) ((const float*)d_in[(i)])
  // pytree (sorted keys): b1a:3..8{b1,b2,g1,g2,w1,w2} b1b:9..14 b2a:15..20 b2b:21..26
  // b3a:27..32 b3b:33..38 bn2_b:39 bn2_g:40 bn3_b:41 bn3_g:42 gn_in_b:43 gn_in_g:44
  // gn_out_b:45 gn_out_g:46 w_ds2:47 w_ds3:48 w_in:49 w_out:50

  // ---- dims (unpadded / padded) ----
  const int D1 = 16, H1 = 200, W1 = 200, Hp1 = 202, Wp1 = 202, Dp1 = 18;
  const int D2 = 8, H2 = 100, W2 = 100, Hp2 = 102, Wp2 = 102, Dp2 = 10;
  const int D3 = 4, H3 = 50, W3 = 50, Hp3 = 52, Wp3 = 52, Dp3 = 6;
  const long P1 = (long)D1 * H1 * W1;
  const long P2 = (long)D2 * H2 * W2;
  const long P3 = (long)D3 * H3 * W3;
  const long Pp1 = (long)Dp1 * Hp1 * Wp1;   // 734472
  const long Pp2 = (long)Dp2 * Hp2 * Wp2;   // 104040
  const long Pp3 = (long)Dp3 * Hp3 * Wp3;   // 16224

  const size_t szA1 = (size_t)Pp1 * 32 * 2;   // 47,006,208
  const size_t szX16 = (size_t)Pp1 * 16 * 2;  // 23,503,104
  const size_t szA2 = (size_t)Pp2 * 64 * 2;   // 13,317,120
  const size_t szA3 = (size_t)Pp3 * 128 * 2;  //  4,153,344

  // ---- workspace layout ----
  char* ws = (char*)d_ws;
  size_t off = 0;
  float* Y = (float*)(ws + off);    off = align256(off + (size_t)P1 * 32 * 4);
  char* A0s1 = ws + off;            off = align256(off + szA1);
  char* A1s1 = ws + off;            off = align256(off + szA1);
  // overlays (lifetimes are disjoint; each zeroed at the right stream point):
  u16* X16 = (u16*)A1s1;
  char* A0s2 = A1s1;
  char* A1s2 = A1s1 + align256(szA2);
  char* A0s3 = A0s1;
  char* A1s3 = A0s1 + align256(szA3);
  float* M1 = (float*)(ws + off);   off = align256(off + (size_t)P1 * 4);
  float* M2 = (float*)(ws + off);   off = align256(off + (size_t)P2 * 4);
  float* M3 = (float*)(ws + off);   off = align256(off + (size_t)P3 * 4);
  float* SUMS = (float*)(ws + off); off = align256(off + 4096);
  auto wpsize = [](int cout, int cin) -> size_t {
    int kc = (27 * cin + 31) / 32;
    return (size_t)(cout / 16) * kc * 512 * 2;
  };
  u16* wpIN = (u16*)(ws + off);  off = align256(off + wpsize(32, 16));
  u16* wp1a1 = (u16*)(ws + off); off = align256(off + wpsize(32, 32));
  u16* wp1a2 = (u16*)(ws + off); off = align256(off + wpsize(32, 32));
  u16* wp1b1 = (u16*)(ws + off); off = align256(off + wpsize(32, 32));
  u16* wp1b2 = (u16*)(ws + off); off = align256(off + wpsize(32, 32));
  u16* wpDS2 = (u16*)(ws + off); off = align256(off + wpsize(64, 32));
  u16* wp2a1 = (u16*)(ws + off); off = align256(off + wpsize(64, 64));
  u16* wp2a2 = (u16*)(ws + off); off = align256(off + wpsize(64, 64));
  u16* wp2b1 = (u16*)(ws + off); off = align256(off + wpsize(64, 64));
  u16* wp2b2 = (u16*)(ws + off); off = align256(off + wpsize(64, 64));
  u16* wpDS3 = (u16*)(ws + off); off = align256(off + wpsize(128, 64));
  u16* wp3a1 = (u16*)(ws + off); off = align256(off + wpsize(128, 128));
  u16* wp3a2 = (u16*)(ws + off); off = align256(off + wpsize(128, 128));
  u16* wp3b1 = (u16*)(ws + off); off = align256(off + wpsize(128, 128));
  u16* wp3b2 = (u16*)(ws + off); off = align256(off + wpsize(128, 128));
  u16* wpOUT = (u16*)(ws + off); off = align256(off + wpsize(128, 128));
  (void)ws_size; (void)n_in; (void)out_size;

  // ---- init: zero A0s1 (full), X16 region, mask; scatter ----
  k_zero<<<4096, 256, 0, stream>>>((uint4*)A0s1, (long)(szA1 / 16));
  k_zero<<<4096, 256, 0, stream>>>((uint4*)X16, (long)(szX16 / 16));
  k_zero<<<1024, 256, 0, stream>>>((uint4*)M1, (long)(P1 * 4 / 16));
  k_scatter<<<(N + 255) / 256, 256, 0, stream>>>(vf, coors, N, X16, M1, H1, W1, Hp1, Wp1);

  // ---- pack all 16 conv weights ----
  run_pack(stream, FP(49), wpIN, 32, 16);
  run_pack(stream, FP(7), wp1a1, 32, 32);
  run_pack(stream, FP(8), wp1a2, 32, 32);
  run_pack(stream, FP(13), wp1b1, 32, 32);
  run_pack(stream, FP(14), wp1b2, 32, 32);
  run_pack(stream, FP(47), wpDS2, 64, 32);
  run_pack(stream, FP(19), wp2a1, 64, 64);
  run_pack(stream, FP(20), wp2a2, 64, 64);
  run_pack(stream, FP(25), wp2b1, 64, 64);
  run_pack(stream, FP(26), wp2b2, 64, 64);
  run_pack(stream, FP(48), wpDS3, 128, 64);
  run_pack(stream, FP(31), wp3a1, 128, 128);
  run_pack(stream, FP(32), wp3a2, 128, 128);
  run_pack(stream, FP(37), wp3b1, 128, 128);
  run_pack(stream, FP(38), wp3b2, 128, 128);
  run_pack(stream, FP(50), wpOUT, 128, 128);

  // ---- stage 1 ----
  run_conv(stream, X16, wpIN, Y, 16, 32, 1, Hp1, Wp1, D1, H1, W1);
  k_gn_apply_bf16<<<(int)((P1 * 16 + 255) / 256), 256, 0, stream>>>(
      Y, FP(44), FP(43), M1, (u16*)A0s1, 32, 2, P1, H1, W1, Hp1, Wp1);
  // A1s1 first write comes next; clear stale X16 bytes + halo now
  k_zero<<<4096, 256, 0, stream>>>((uint4*)A1s1, (long)(szA1 / 16));

  // b1a
  run_conv(stream, (u16*)A0s1, wp1a1, Y, 32, 32, 1, Hp1, Wp1, D1, H1, W1);
  run_bn(stream, Y, SUMS, FP(5), FP(3), M1, nullptr, (u16*)A1s1, 32, D1, H1, W1, Hp1, Wp1);
  run_conv(stream, (u16*)A1s1, wp1a2, Y, 32, 32, 1, Hp1, Wp1, D1, H1, W1);
  run_bn(stream, Y, SUMS, FP(6), FP(4), M1, (u16*)A0s1, (u16*)A0s1, 32, D1, H1, W1, Hp1, Wp1);
  // b1b
  run_conv(stream, (u16*)A0s1, wp1b1, Y, 32, 32, 1, Hp1, Wp1, D1, H1, W1);
  run_bn(stream, Y, SUMS, FP(11), FP(9), M1, nullptr, (u16*)A1s1, 32, D1, H1, W1, Hp1, Wp1);
  run_conv(stream, (u16*)A1s1, wp1b2, Y, 32, 32, 1, Hp1, Wp1, D1, H1, W1);
  run_bn(stream, Y, SUMS, FP(12), FP(10), M1, (u16*)A0s1, (u16*)A0s1, 32, D1, H1, W1, Hp1, Wp1);

  // ---- stage 2 (A1s1 dead after b1b conv2 -> carve A0s2/A1s2 out of it) ----
  k_down_mask<<<(int)((P2 + 255) / 256), 256, 0, stream>>>(M1, M2, D1, H1, W1, D2, H2, W2);
  k_zero<<<2048, 256, 0, stream>>>((uint4*)A0s2, (long)(szA2 / 16));
  k_zero<<<2048, 256, 0, stream>>>((uint4*)A1s2, (long)(szA2 / 16));
  run_conv(stream, (u16*)A0s1, wpDS2, Y, 32, 64, 2, Hp1, Wp1, D2, H2, W2);
  // A0s1 dead after ds2 conv -> carve A0s3/A1s3 out of it
  k_zero<<<1024, 256, 0, stream>>>((uint4*)A0s3, (long)(szA3 / 16));
  k_zero<<<1024, 256, 0, stream>>>((uint4*)A1s3, (long)(szA3 / 16));
  run_bn(stream, Y, SUMS, FP(40), FP(39), M2, nullptr, (u16*)A0s2, 64, D2, H2, W2, Hp2, Wp2);
  // b2a
  run_conv(stream, (u16*)A0s2, wp2a1, Y, 64, 64, 1, Hp2, Wp2, D2, H2, W2);
  run_bn(stream, Y, SUMS, FP(17), FP(15), M2, nullptr, (u16*)A1s2, 64, D2, H2, W2, Hp2, Wp2);
  run_conv(stream, (u16*)A1s2, wp2a2, Y, 64, 64, 1, Hp2, Wp2, D2, H2, W2);
  run_bn(stream, Y, SUMS, FP(18), FP(16), M2, (u16*)A0s2, (u16*)A0s2, 64, D2, H2, W2, Hp2, Wp2);
  // b2b
  run_conv(stream, (u16*)A0s2, wp2b1, Y, 64, 64, 1, Hp2, Wp2, D2, H2, W2);
  run_bn(stream, Y, SUMS, FP(23), FP(21), M2, nullptr, (u16*)A1s2, 64, D2, H2, W2, Hp2, Wp2);
  run_conv(stream, (u16*)A1s2, wp2b2, Y, 64, 64, 1, Hp2, Wp2, D2, H2, W2);
  run_bn(stream, Y, SUMS, FP(24), FP(22), M2, (u16*)A0s2, (u16*)A0s2, 64, D2, H2, W2, Hp2, Wp2);

  // ---- stage 3 ----
  k_down_mask<<<(int)((P3 + 255) / 256), 256, 0, stream>>>(M2, M3, D2, H2, W2, D3, H3, W3);
  run_conv(stream, (u16*)A0s2, wpDS3, Y, 64, 128, 2, Hp2, Wp2, D3, H3, W3);
  run_bn(stream, Y, SUMS, FP(42), FP(41), M3, nullptr, (u16*)A0s3, 128, D3, H3, W3, Hp3, Wp3);
  // b3a
  run_conv(stream, (u16*)A0s3, wp3a1, Y, 128, 128, 1, Hp3, Wp3, D3, H3, W3);
  run_bn(stream, Y, SUMS, FP(29), FP(27), M3, nullptr, (u16*)A1s3, 128, D3, H3, W3, Hp3, Wp3);
  run_conv(stream, (u16*)A1s3, wp3a2, Y, 128, 128, 1, Hp3, Wp3, D3, H3, W3);
  run_bn(stream, Y, SUMS, FP(30), FP(28), M3, (u16*)A0s3, (u16*)A0s3, 128, D3, H3, W3, Hp3, Wp3);
  // b3b
  run_conv(stream, (u16*)A0s3, wp3b1, Y, 128, 128, 1, Hp3, Wp3, D3, H3, W3);
  run_bn(stream, Y, SUMS, FP(35), FP(33), M3, nullptr, (u16*)A1s3, 128, D3, H3, W3, Hp3, Wp3);
  run_conv(stream, (u16*)A1s3, wp3b2, Y, 128, 128, 1, Hp3, Wp3, D3, H3, W3);
  run_bn(stream, Y, SUMS, FP(36), FP(34), M3, (u16*)A0s3, (u16*)A0s3, 128, D3, H3, W3, Hp3, Wp3);

  // ---- conv_out + GN + ReLU + transposed write ----
  run_conv(stream, (u16*)A0s3, wpOUT, Y, 128, 128, 1, Hp3, Wp3, D3, H3, W3);
  k_gn_out<<<(int)((P3 * 16 + 255) / 256), 256, 0, stream>>>(
      Y, FP(46), FP(45), M3, (float*)d_out, D3, H3, W3);
#undef FP
}
